// ResidualTRNN_23467701305953
// MI455X (gfx1250) — compile-verified, matrix operand order checked by reading, not run on hardware
//
#include <hip/hip_runtime.h>
#include <cmath>

#define B_SZ 256
#define T_SZ 2048
#define D_IN 64
#define D_H  512

// padded LDS strides (in f16 elements) to avoid bank conflicts on ds_load_b128
#define A_STRIDE 520   // 520*2B = 1040B -> row-to-row bank offset of 4
#define X_STRIDE 72

typedef _Float16 h8   __attribute__((ext_vector_type(8)));
typedef _Float16 v16h __attribute__((ext_vector_type(16)));
typedef float    v8f  __attribute__((ext_vector_type(8)));

// Assemble a 16xf16 WMMA operand from two 16-byte chunks at p and p+16 elements.
__device__ __forceinline__ v16h load_op(const _Float16* p) {
    h8 lo = *(const h8*)(p);
    h8 hi = *(const h8*)(p + 16);
    return __builtin_shufflevector(lo, hi, 0,1,2,3,4,5,6,7,8,9,10,11,12,13,14,15);
}

#define WMMA_F16(a, b, c) \
    __builtin_amdgcn_wmma_f32_16x16x32_f16(false, (a), false, (b), (short)0, (c), false, false)

__global__ void convert_weights(const float* __restrict__ Wh,
                                const float* __restrict__ Wx,
                                _Float16* __restrict__ whf,
                                _Float16* __restrict__ wxf) {
    int i = blockIdx.x * blockDim.x + threadIdx.x;
    if (i < D_H * D_H) whf[i] = (_Float16)Wh[i];
    if (i < D_H * D_IN) wxf[i] = (_Float16)Wx[i];
}

__global__ __launch_bounds__(512)
void trnn_scan_kernel(const float* __restrict__ x,
                      const float* __restrict__ Wx_b,
                      const float* __restrict__ decay,
                      const float* __restrict__ out_w,
                      const float* __restrict__ out_b,
                      const _Float16* __restrict__ whf,
                      const _Float16* __restrict__ wxf,
                      float* __restrict__ out) {
    __shared__ float     hbuf[16 * D_H];        // fp32 hidden state, 16 batch rows
    __shared__ _Float16  a16[16 * A_STRIDE];    // f16 (h * gamma)  -> WMMA A operand
    __shared__ _Float16  xa16[16 * X_STRIDE];   // f16 x_t tile     -> WMMA A operand
    __shared__ float     adec[D_H];
    __shared__ float     outw[D_H];
    __shared__ float     wxb[D_H];

    const int tid  = threadIdx.x;
    const int b0   = blockIdx.x * 16;           // batch tile base

    // one-time loads
    if (tid < D_H) {
        adec[tid] = fabsf(decay[tid]);
        outw[tid] = out_w[tid];
        wxb[tid]  = Wx_b[tid];
    }
    for (int i = tid; i < 16 * D_H; i += 512) hbuf[i] = 0.0f;
    const float outb = out_b[0];

    // ---- per-wave WMMA geometry (wave32) ----
    const int wv   = tid >> 5;                  // wave 0..15, owns 32 output cols
    const int lane = tid & 31;
    const int nn   = lane & 15;                 // N (or M) within tile
    const int kb8  = (lane >> 4) << 3;          // K sub-offset: 0 / 8
    const int mhi  = (lane >> 4) << 3;          // D-matrix row offset: 0 / 8
    const int n0   = wv * 32;

    const _Float16* aB = &a16 [nn * A_STRIDE + kb8];
    const _Float16* xB = &xa16[nn * X_STRIDE + kb8];
    // B operand = rows of Wh / Wx (row-major), lane n -> row n0+nn(+16)
    const _Float16* rowW0 = whf + (size_t)(n0 + nn) * D_H  + kb8;
    const _Float16* rowW1 = rowW0 + 16 * D_H;
    const _Float16* rowX0 = wxf + (size_t)(n0 + nn) * D_IN + kb8;
    const _Float16* rowX1 = rowX0 + 16 * D_IN;

    // ---- persistent operands in registers ----
    // Wx: 2 K-blocks x 2 tiles = 32 VGPRs.  Wh low half (K=0..255): 8 x 2 x 8 = 128 VGPRs.
    v16h wxr0[2], wxr1[2];
#pragma unroll
    for (int kb = 0; kb < 2; ++kb) {
        wxr0[kb] = load_op(rowX0 + kb * 32);
        wxr1[kb] = load_op(rowX1 + kb * 32);
    }
    v16h wh0[8], wh1[8];
#pragma unroll
    for (int kb = 0; kb < 8; ++kb) {
        wh0[kb] = load_op(rowW0 + kb * 32);
        wh1[kb] = load_op(rowW1 + kb * 32);
    }

    // phase-1 indices: one batch row per wave
    const int    m1 = tid >> 5;                 // 0..15
    const int    kc = lane * 16;                // 16 contiguous k's per thread
    const int    d0 = lane * 2;
    const size_t xrow_base = (size_t)(b0 + m1) * T_SZ * D_IN;

    __syncthreads();

    for (int t = 0; t < T_SZ; ++t) {
        // ---------- phase 1: build f16 A operands in LDS ----------
        const float* xr = x + xrow_base + (size_t)t * D_IN;
        const float dtm = xr[2];
        {
            float2 xv = *(const float2*)(xr + d0);
            xa16[m1 * X_STRIDE + d0]     = (_Float16)xv.x;
            xa16[m1 * X_STRIDE + d0 + 1] = (_Float16)xv.y;
        }
#pragma unroll
        for (int i = 0; i < 16; ++i) {
            const int k = kc + i;
            const float g = __expf(-dtm * adec[k]);
            a16[m1 * A_STRIDE + k] = (_Float16)(hbuf[m1 * D_H + k] * g);
        }
        if (t + 1 < T_SZ) __builtin_prefetch(xr + D_IN, 0, 0);  // global_prefetch_b8
        __syncthreads();

        // ---------- phase 2: WMMA  acc = x_t@Wx^T + (h*gamma)@Wh^T ----------
        v8f acc0 = {0.f,0.f,0.f,0.f,0.f,0.f,0.f,0.f};
        v8f acc1 = acc0;
        // input GEMM: B operands fully register-resident
        {
            v16h ax0 = load_op(xB);
            v16h ax1 = load_op(xB + 32);
            acc0 = WMMA_F16(ax0, wxr0[0], acc0);
            acc1 = WMMA_F16(ax0, wxr1[0], acc1);
            acc0 = WMMA_F16(ax1, wxr0[1], acc0);
            acc1 = WMMA_F16(ax1, wxr1[1], acc1);
        }
        // recurrent GEMM over 16 K-blocks:
        //   i = 0..7  : register-resident Wh (low half of K)
        //   i = 8..15 : Wh streamed from L2 with t-rotated K-block order (defeats
        //               LICM so the loads stay in-loop instead of spilling)
        // A operand is explicitly double-buffered so the next ds_load_b128 pair can
        // issue while the current WMMAs consume abuf[i&1] (partial dscnt waits).
        v16h abuf[2];
        abuf[0] = load_op(aB);
#pragma unroll
        for (int i = 0; i < 16; ++i) {
            if (i < 15) {
                const int knext = (i + 1 < 8) ? (i + 1) * 32
                                              : (8 + ((i + 1 - 8 + t) & 7)) * 32;
                abuf[(i + 1) & 1] = load_op(aB + knext);
            }
            if (i < 8) {
                acc0 = WMMA_F16(abuf[i & 1], wh0[i], acc0);
                acc1 = WMMA_F16(abuf[i & 1], wh1[i], acc1);
            } else {
                const int koff = (8 + ((i - 8 + t) & 7)) * 32;
                v16h b0 = load_op(rowW0 + koff);
                v16h b1 = load_op(rowW1 + koff);
                acc0 = WMMA_F16(abuf[i & 1], b0, acc0);
                acc1 = WMMA_F16(abuf[i & 1], b1, acc1);
            }
        }
        // writeback: D[r][lane] -> M = r + mhi, N = nn ; h = tanh(acc + Wx_b)
        {
            const int nA = n0 + nn;
            const int nB = nA + 16;
            const float biasA = wxb[nA];
            const float biasB = wxb[nB];
#pragma unroll
            for (int r = 0; r < 8; ++r) {
                hbuf[(r + mhi) * D_H + nA] = tanhf(acc0[r] + biasA);
                hbuf[(r + mhi) * D_H + nB] = tanhf(acc1[r] + biasB);
            }
        }
        __syncthreads();

        // ---------- phase 3: output projection r = 0.3*tanh(h.out_w + b) ----------
        float s = 0.0f;
#pragma unroll
        for (int i = 0; i < 16; ++i) {
            const int k = lane + 32 * i;
            s += hbuf[m1 * D_H + k] * outw[k];
        }
#pragma unroll
        for (int off = 16; off >= 1; off >>= 1) s += __shfl_xor(s, off, 32);
        if (lane == 0) out[(size_t)(b0 + m1) * T_SZ + t] = 0.3f * tanhf(s + outb);
    }
}

extern "C" void kernel_launch(void* const* d_in, const int* in_sizes, int n_in,
                              void* d_out, int out_size, void* d_ws, size_t ws_size,
                              hipStream_t stream) {
    const float* x     = (const float*)d_in[0];
    const float* Wx_w  = (const float*)d_in[1];
    const float* Wx_b  = (const float*)d_in[2];
    const float* Wh_w  = (const float*)d_in[3];
    const float* decay = (const float*)d_in[4];
    const float* out_w = (const float*)d_in[5];
    const float* out_b = (const float*)d_in[6];
    float* out = (float*)d_out;

    _Float16* whf = (_Float16*)d_ws;            // 512*512 f16 = 512 KB
    _Float16* wxf = whf + D_H * D_H;            // 512*64  f16 =  64 KB

    convert_weights<<<(D_H * D_H + 255) / 256, 256, 0, stream>>>(Wh_w, Wx_w, whf, wxf);
    trnn_scan_kernel<<<B_SZ / 16, 512, 0, stream>>>(x, Wx_b, decay, out_w, out_b,
                                                    whf, wxf, out);
}
